// RGCNEncoder_3616362463539
// MI455X (gfx1250) — matrix operand order repaired and checked by reading, not compiled, not run on hardware
//
#include <hip/hip_runtime.h>
#include <hip/hip_bf16.h>

typedef __attribute__((ext_vector_type(16))) __bf16 v16bf;
typedef __attribute__((ext_vector_type(8)))  __bf16 v8bf;
typedef __attribute__((ext_vector_type(8)))  float  v8f;

#define HD 64
#define NREL 50

// ---------------------------------------------------------------------------
// degree counts: cnt[n*NREL + r] = #edges with dst==n, etype==r (float)
// ---------------------------------------------------------------------------
__global__ __launch_bounds__(256)
void zero_f32(float* __restrict__ p, long long n) {
  long long i = (long long)blockIdx.x * blockDim.x + threadIdx.x;
  if (i < n) p[i] = 0.0f;
}

__global__ __launch_bounds__(256)
void count_edges(const int* __restrict__ dst, const int* __restrict__ etype,
                 float* __restrict__ cnt, int n_edges) {
  int e = blockIdx.x * blockDim.x + threadIdx.x;
  if (e < n_edges)
    atomicAdd(&cnt[(long long)dst[e] * NREL + etype[e]], 1.0f);
}

// ---------------------------------------------------------------------------
// f32 -> bf16 shadow copy (layer-0 input)
// ---------------------------------------------------------------------------
__global__ __launch_bounds__(256)
void conv_bf16(const float* __restrict__ src, __bf16* __restrict__ dstb, long long n) {
  long long i = (long long)blockIdx.x * blockDim.x + threadIdx.x;
  if (i < n) dstb[i] = (__bf16)src[i];
}

// ---------------------------------------------------------------------------
// ReLU in place (f32) + bf16 shadow copy for the next WMMA pass
// ---------------------------------------------------------------------------
__global__ __launch_bounds__(256)
void relu_fused(float* __restrict__ p, __bf16* __restrict__ pb, long long n) {
  long long i = (long long)blockIdx.x * blockDim.x + threadIdx.x;
  if (i < n) {
    float v = fmaxf(p[i], 0.0f);
    p[i]  = v;
    pb[i] = (__bf16)v;
  }
}

// ---------------------------------------------------------------------------
// Pack w_root[k][col] (f32, 64x64) into bf16 B-operand lane layout.
// Flat idx = ((c*4 + ntile)*32 + lane)*16 + i ;  k = c*32 + (i>>3)*16 + (lane>>4)*8 + (i&7),
// col = ntile*16 + (lane&15).  4096 elements total.
// ---------------------------------------------------------------------------
__global__ __launch_bounds__(256)
void pack_w_bf16(const float* __restrict__ w, __bf16* __restrict__ wb) {
  int idx = blockIdx.x * blockDim.x + threadIdx.x;   // 0..4095
  if (idx >= 2 * 4 * 32 * 16) return;
  const int i     = idx & 15;
  const int lane  = (idx >> 4) & 31;
  const int ntile = (idx >> 9) & 3;
  const int c     = idx >> 11;
  const int k     = c * 32 + (i >> 3) * 16 + (lane >> 4) * 8 + (i & 7);
  const int col   = ntile * 16 + (lane & 15);
  wb[idx] = (__bf16)w[k * HD + col];
}

// ---------------------------------------------------------------------------
// Root transform over FULL 16-row tiles: out = h @ w_root + bias (WMMA bf16).
// blockDim=128: 4 waves, wave `ntile` owns output cols [16*ntile, 16*ntile+16).
// A from bf16 activations (4x b128 loads/lane), B pre-packed (2x 32B loads).
// D staged via padded LDS -> coalesced float4 stores.
// ---------------------------------------------------------------------------
__global__ __launch_bounds__(128)
void root_wmma(const __bf16* __restrict__ hb, const __bf16* __restrict__ wb,
               const float* __restrict__ bias, float* __restrict__ out) {
  __shared__ float tile[16][HD + 1];
  const int lane  = threadIdx.x & 31;
  const int ntile = threadIdx.x >> 5;
  const int row16 = lane & 15;
  const int half  = lane >> 4;
  const int mbase = blockIdx.x * 16;

  // ---- A operand (16x32 bf16 per chunk), ISA 7.12.2 layout ----
  const __bf16* xr = hb + (long long)(mbase + row16) * HD;
  const v8bf alo0 = *(const v8bf*)(xr + half * 8);
  const v8bf ahi0 = *(const v8bf*)(xr + 16 + half * 8);
  const v8bf alo1 = *(const v8bf*)(xr + 32 + half * 8);
  const v8bf ahi1 = *(const v8bf*)(xr + 48 + half * 8);
  const v16bf a0 = __builtin_shufflevector(alo0, ahi0, 0,1,2,3,4,5,6,7,8,9,10,11,12,13,14,15);
  const v16bf a1 = __builtin_shufflevector(alo1, ahi1, 0,1,2,3,4,5,6,7,8,9,10,11,12,13,14,15);

  // ---- B operand: pre-packed lane layout ----
  const v16bf b0 = *(const v16bf*)(wb + ((0 * 4 + ntile) * 32 + lane) * 16);
  const v16bf b1 = *(const v16bf*)(wb + ((1 * 4 + ntile) * 32 + lane) * 16);

  v8f c = {};
  c = __builtin_amdgcn_wmma_f32_16x16x32_bf16(false, a0, false, b0, (short)0, c, false, false);
  c = __builtin_amdgcn_wmma_f32_16x16x32_bf16(false, a1, false, b1, (short)0, c, false, false);

  // ---- stage D (+bias) into LDS; elem i -> row half*8+i, col lane&15 ----
  const int col = ntile * 16 + row16;
  const float bj = bias[col];
#pragma unroll
  for (int i = 0; i < 8; ++i)
    tile[half * 8 + i][col] = c[i] + bj;
  __syncthreads();

  // ---- coalesced store: 128 threads, 16 rows x 64 cols, float4 x2 each ----
  const int row = threadIdx.x >> 3;
  const int c0  = (threadIdx.x & 7) * 8;
  float* orow = out + (long long)(mbase + row) * HD;
  float4 va, vb2;
  va.x  = tile[row][c0 + 0]; va.y  = tile[row][c0 + 1];
  va.z  = tile[row][c0 + 2]; va.w  = tile[row][c0 + 3];
  vb2.x = tile[row][c0 + 4]; vb2.y = tile[row][c0 + 5];
  vb2.z = tile[row][c0 + 6]; vb2.w = tile[row][c0 + 7];
  *(float4*)(orow + c0)     = va;
  *(float4*)(orow + c0 + 4) = vb2;
}

// Scalar tail for rows not covered by full tiles (launched only if needed).
__global__ __launch_bounds__(256)
void root_tail(const float* __restrict__ h, const float* __restrict__ w,
               const float* __restrict__ bias, float* __restrict__ out,
               int mstart, int n_nodes) {
  int idx = blockIdx.x * blockDim.x + threadIdx.x;
  int m = mstart + (idx >> 6);
  int col = idx & 63;
  if (m >= n_nodes) return;
  const float* hr = h + (long long)m * HD;
  float acc = bias[col];
  for (int k = 0; k < HD; ++k) acc += hr[k] * w[k * HD + col];
  out[(long long)m * HD + col] = acc;
}

// ---------------------------------------------------------------------------
// Block-diagonal relation transform + mean-scaled scatter.
// 4 edges / 256-thread block; 64 threads per edge. w_rel = [R,4,16,16].
// ---------------------------------------------------------------------------
__global__ __launch_bounds__(256)
void edge_blockdiag(const float* __restrict__ h_in, const int* __restrict__ src,
                    const int* __restrict__ dst, const int* __restrict__ etype,
                    const float* __restrict__ w_rel, const float* __restrict__ cnt,
                    float* __restrict__ h_out, int n_edges) {
  __shared__ float xs[4][HD];
  const int eg = threadIdx.x >> 6;
  const int f  = threadIdx.x & 63;
  const int e  = blockIdx.x * 4 + eg;
  const bool live = (e < n_edges);

  const int s = live ? src[e] : 0;
  xs[eg][f] = h_in[(long long)s * HD + f];
  __syncthreads();

  if (!live) return;
  const int r  = etype[e];
  const int d0 = dst[e];
  const int b = f >> 4, j = f & 15;
  const float* wp = w_rel + (((long long)r * 4 + b) * 16) * 16 + j;  // stride 16 per d
  const float* xb = &xs[eg][b * 16];
  float acc = 0.0f;
#pragma unroll
  for (int d = 0; d < 16; ++d) acc += xb[d] * wp[d * 16];

  const float cdeg = cnt[(long long)d0 * NREL + r];
  atomicAdd(&h_out[(long long)d0 * HD + f], acc / fmaxf(cdeg, 1.0f));
}

// ---------------------------------------------------------------------------
// Full per-relation transform (final layer): w2[r][d][f]
// ---------------------------------------------------------------------------
__global__ __launch_bounds__(256)
void edge_full(const float* __restrict__ h_in, const int* __restrict__ src,
               const int* __restrict__ dst, const int* __restrict__ etype,
               const float* __restrict__ w_rel, const float* __restrict__ cnt,
               float* __restrict__ h_out, int n_edges) {
  __shared__ float xs[4][HD];
  const int eg = threadIdx.x >> 6;
  const int f  = threadIdx.x & 63;
  const int e  = blockIdx.x * 4 + eg;
  const bool live = (e < n_edges);

  const int s = live ? src[e] : 0;
  xs[eg][f] = h_in[(long long)s * HD + f];
  __syncthreads();

  if (!live) return;
  const int r  = etype[e];
  const int d0 = dst[e];
  const float* wp = w_rel + ((long long)r * HD) * HD + f;            // stride 64 per d
  float acc = 0.0f;
#pragma unroll
  for (int d = 0; d < HD; ++d) acc += xs[eg][d] * wp[(long long)d * HD];

  const float cdeg = cnt[(long long)d0 * NREL + r];
  atomicAdd(&h_out[(long long)d0 * HD + f], acc / fmaxf(cdeg, 1.0f));
}

// ---------------------------------------------------------------------------
extern "C" void kernel_launch(void* const* d_in, const int* in_sizes, int n_in,
                              void* d_out, int out_size, void* d_ws, size_t ws_size,
                              hipStream_t stream) {
  (void)n_in; (void)out_size; (void)ws_size;

  const float* x       = (const float*)d_in[0];
  const int*   eidx    = (const int*)  d_in[1];
  const int*   etype   = (const int*)  d_in[2];
  const float* w0_rel  = (const float*)d_in[3];
  const float* w0_root = (const float*)d_in[4];
  const float* b0      = (const float*)d_in[5];
  const float* w1_rel  = (const float*)d_in[6];
  const float* w1_root = (const float*)d_in[7];
  const float* b1      = (const float*)d_in[8];
  const float* w2_rel  = (const float*)d_in[9];
  const float* w2_root = (const float*)d_in[10];
  const float* b2      = (const float*)d_in[11];
  float*       out     = (float*)d_out;

  const int n_edges = in_sizes[2];
  const int n_nodes = in_sizes[0] / HD;
  const int* src = eidx;
  const int* dst = eidx + n_edges;

  // ---- workspace carve-up (64B-aligned sections) ----
  char* wsb = (char*)d_ws;
  auto align64 = [](size_t v) { return (v + 63) & ~(size_t)63; };
  size_t off = 0;
  float* cnt = (float*)(wsb + off); off = align64(off + (size_t)n_nodes * NREL * 4);
  float* h0  = (float*)(wsb + off); off = align64(off + (size_t)n_nodes * HD * 4);
  float* h1  = (float*)(wsb + off); off = align64(off + (size_t)n_nodes * HD * 4);
  __bf16* hb = (__bf16*)(wsb + off); off = align64(off + (size_t)n_nodes * HD * 2);
  __bf16* wb = (__bf16*)(wsb + off); off = align64(off + (size_t)4096 * 2);

  const long long ncnt = (long long)n_nodes * NREL;
  const long long nh   = (long long)n_nodes * HD;
  const int edge_blocks = (n_edges + 3) / 4;
  const int n_tiles = n_nodes / 16;                 // full tiles (N=50000 -> 3125, rem 0)
  const int mrem    = n_nodes - n_tiles * 16;
  const int grid_nh = (int)((nh + 255) / 256);

  // degree counts (shared by all three layers)
  zero_f32<<<(int)((ncnt + 255) / 256), 256, 0, stream>>>(cnt, ncnt);
  count_edges<<<(n_edges + 255) / 256, 256, 0, stream>>>(dst, etype, cnt, n_edges);

  // ---- layer 0 ----
  conv_bf16<<<grid_nh, 256, 0, stream>>>(x, hb, nh);
  pack_w_bf16<<<16, 256, 0, stream>>>(w0_root, wb);
  if (n_tiles) root_wmma<<<n_tiles, 128, 0, stream>>>(hb, wb, b0, h0);
  if (mrem) root_tail<<<(mrem * HD + 255) / 256, 256, 0, stream>>>(x, w0_root, b0, h0, n_tiles * 16, n_nodes);
  edge_blockdiag<<<edge_blocks, 256, 0, stream>>>(x, src, dst, etype, w0_rel, cnt, h0, n_edges);
  relu_fused<<<grid_nh, 256, 0, stream>>>(h0, hb, nh);

  // ---- layer 1 ----
  pack_w_bf16<<<16, 256, 0, stream>>>(w1_root, wb);
  if (n_tiles) root_wmma<<<n_tiles, 128, 0, stream>>>(hb, wb, b1, h1);
  if (mrem) root_tail<<<(mrem * HD + 255) / 256, 256, 0, stream>>>(h0, w1_root, b1, h1, n_tiles * 16, n_nodes);
  edge_blockdiag<<<edge_blocks, 256, 0, stream>>>(h0, src, dst, etype, w1_rel, cnt, h1, n_edges);
  relu_fused<<<grid_nh, 256, 0, stream>>>(h1, hb, nh);

  // ---- layer 2 (full relation weights, no relu) ----
  pack_w_bf16<<<16, 256, 0, stream>>>(w2_root, wb);
  if (n_tiles) root_wmma<<<n_tiles, 128, 0, stream>>>(hb, wb, b2, out);
  if (mrem) root_tail<<<(mrem * HD + 255) / 256, 256, 0, stream>>>(h1, w2_root, b2, out, n_tiles * 16, n_nodes);
  edge_full<<<edge_blocks, 256, 0, stream>>>(h1, src, dst, etype, w2_rel, cnt, out, n_edges);
}
